// GraphFP_72739566125920
// MI455X (gfx1250) — matrix operand order repaired and checked by reading, not compiled, not run on hardware
//
#include <hip/hip_runtime.h>
#include <hip/hip_bf16.h>

typedef _Float16 v16h __attribute__((ext_vector_type(16)));
typedef _Float16 v8h  __attribute__((ext_vector_type(8)));
typedef float    v8f  __attribute__((ext_vector_type(8)));

__device__ __forceinline__ v16h cat8(v8h a, v8h b) {
    return __builtin_shufflevector(a, b, 0,1,2,3,4,5,6,7,8,9,10,11,12,13,14,15);
}

// ---------------------------------------------------------------------------
// K0: zero the fp partial accumulator (workspace is poisoned by harness)
// ---------------------------------------------------------------------------
__global__ void zero_kernel(float* __restrict__ p, int nelem) {
    int i = blockIdx.x * 256 + threadIdx.x;
    if (i < nelem) p[i] = 0.f;
}

// ---------------------------------------------------------------------------
// K1: W_output (4,39,2048) f32  ->  woutT (4,2048,64) f16, K padded to 64 w/ 0
// ---------------------------------------------------------------------------
__global__ __launch_bounds__(256) void woutT_kernel(const float* __restrict__ W_output,
                                                    _Float16* __restrict__ woutT) {
    int idx = blockIdx.x * 256 + threadIdx.x;          // [0, 4*2048*64)
    if (idx >= 4 * 2048 * 64) return;
    int k = idx & 63;
    int o = (idx >> 6) & 2047;
    int d = idx >> 17;
    woutT[idx] = (k < 39) ? (_Float16)W_output[((size_t)d * 39 + k) * 2048 + o]
                          : (_Float16)0.f;
}

// ---------------------------------------------------------------------------
// K2: per (b,i): S0[f] = sum_j x[b,i,j,f]; attrs = x[b,i,i,:] (f32 + padded f16)
// One block (64 threads) per (b,i); thread j loads its 40-float row via b128.
// ---------------------------------------------------------------------------
__global__ __launch_bounds__(64) void prep_kernel(const float* __restrict__ x,
                                                  float* __restrict__ attrs32,
                                                  _Float16* __restrict__ attrs16,
                                                  float* __restrict__ S) {
    __shared__ float rows[64 * 40];
    const int t = threadIdx.x;
    const size_t base = (size_t)blockIdx.x * (64 * 40);
    const float4* p = (const float4*)(x + base + t * 40);
    float4* q = (float4*)(rows + t * 40);
#pragma unroll
    for (int k = 0; k < 10; ++k) q[k] = p[k];
    __syncthreads();
    const int i = blockIdx.x & 63;  // atom index
    if (t < 40) {
        float s = 0.f;
        for (int j = 0; j < 64; ++j) s += rows[j * 40 + t];
        S[(size_t)blockIdx.x * 40 + t] = s;
        float a = rows[i * 40 + t];
        attrs32[(size_t)blockIdx.x * 40 + t] = a;
        attrs16[(size_t)blockIdx.x * 64 + t] = (_Float16)a;
    } else {
        attrs16[(size_t)blockIdx.x * 64 + t] = (_Float16)0.f;  // K pad
    }
}

// ---------------------------------------------------------------------------
// K3: c[b,i] = 1 + sum_j x[b,j,i,39]   (column sum of bond channel)
// ---------------------------------------------------------------------------
__global__ __launch_bounds__(64) void colsum_kernel(const float* __restrict__ x,
                                                    float* __restrict__ c) {
    const int b = blockIdx.x;
    const int i = threadIdx.x;
    const float* xb = x + (size_t)b * (64 * 64 * 40);
    float s = 1.f;
    for (int j = 0; j < 64; ++j) s += xb[(size_t)(j * 64 + i) * 40 + 39];
    c[b * 64 + i] = s;
}

// ---------------------------------------------------------------------------
// K4: depth update d: new = S@W_inner[d]+b_inner[d]; delta=new-attrs;
//     S += c*delta; attrs = new (f32 + f16). 4 rows per 256-thread block.
// ---------------------------------------------------------------------------
__global__ __launch_bounds__(256) void depth_kernel(const float* __restrict__ W_inner,
                                                    const float* __restrict__ b_inner,
                                                    float* __restrict__ S,
                                                    float* __restrict__ attrs32,
                                                    _Float16* __restrict__ attrs16,
                                                    const float* __restrict__ c, int d) {
    __shared__ float Srow[4][40];
    const int rg = threadIdx.x >> 6;
    const int lt = threadIdx.x & 63;
    const size_t r = (size_t)blockIdx.x * 4 + rg;   // row in [0, 512*64)
    if (lt < 40) Srow[rg][lt] = S[r * 40 + lt];
    __syncthreads();
    if (lt < 39) {
        const int g = lt;
        const float* Wd = W_inner + (size_t)d * 39 * 39;
        float acc = b_inner[d * 39 + g];
        for (int f = 0; f < 39; ++f) acc += Srow[rg][f] * Wd[f * 39 + g];
        const float a_old = attrs32[r * 40 + g];
        const float delta = acc - a_old;
        S[r * 40 + g] = Srow[rg][g] + c[r] * delta;
        attrs32[r * 40 + g] = acc;
        attrs16[r * 64 + g] = (_Float16)acc;
    }
}

// ---------------------------------------------------------------------------
// K5: fingerprint contribution for depth d.
// Block = (b, half): 32 atoms x 2048 cols. 8 waves: wrow(2) x wcol(4).
// Wave: 16 rows x 512 cols in registers (32 v8f accumulators), 64 WMMAs.
// Softmax over o (full 2048 in-block), then race-free partial-fp reduction.
// ---------------------------------------------------------------------------
__global__ __launch_bounds__(256) void fp_kernel(const _Float16* __restrict__ attrs16,
                                                 const _Float16* __restrict__ woutT,
                                                 const float* __restrict__ bout,
                                                 float* __restrict__ fppart, int d) {
    __shared__ float red[32][64];
    __shared__ float rowmax[32];
    __shared__ float rowsum[32];
    __shared__ float fpred[2048][4];

    const int tid  = threadIdx.x;
    const int lane = tid & 31;
    const int w    = tid >> 5;   // wave 0..7
    const int wrow = w >> 2;     // 0..1
    const int wcol = w & 3;      // 0..3
    const int n    = lane & 15;
    const int hi   = lane >> 4;

    const int b  = blockIdx.x >> 1;
    const int hb = blockIdx.x & 1;
    const int m0 = hb * 32 + wrow * 16;

    // A fragments: 16 rows x K64 (two K32 chunks), ISA 16-bit A layout.
    const _Float16* arow = attrs16 + ((size_t)b * 64 + (m0 + n)) * 64;
    v16h a0 = cat8(*(const v8h*)(arow + 0 * 32 + hi * 8),
                   *(const v8h*)(arow + 0 * 32 + 16 + hi * 8));
    v16h a1 = cat8(*(const v8h*)(arow + 1 * 32 + hi * 8),
                   *(const v8h*)(arow + 1 * 32 + 16 + hi * 8));

    const _Float16* wt = woutT + (size_t)d * (2048 * 64);
    const float*    bo = bout + (size_t)d * 2048;

    v8f acc[32];
#pragma unroll
    for (int t = 0; t < 32; ++t) {
        const int c = wcol * 512 + t * 16 + n;
        const _Float16* wc = wt + (size_t)c * 64;     // K-contiguous (transposed)
        v16h b0 = cat8(*(const v8h*)(wc + 0 * 32 + hi * 8),
                       *(const v8h*)(wc + 0 * 32 + 16 + hi * 8));
        v16h b1 = cat8(*(const v8h*)(wc + 1 * 32 + hi * 8),
                       *(const v8h*)(wc + 1 * 32 + 16 + hi * 8));
        v8f dacc = {};
        dacc = __builtin_amdgcn_wmma_f32_16x16x32_f16(false, a0, false, b0,
                                                      (short)0, dacc, false, false);
        dacc = __builtin_amdgcn_wmma_f32_16x16x32_f16(false, a1, false, b1,
                                                      (short)0, dacc, false, false);
        const float bias = bo[c];
#pragma unroll
        for (int v = 0; v < 8; ++v) dacc[v] += bias;
        acc[t] = dacc;
    }

    // --- row max over all 2048 columns ---
    float pm[8];
#pragma unroll
    for (int v = 0; v < 8; ++v) pm[v] = -3.402823466e+38f;
#pragma unroll
    for (int t = 0; t < 32; ++t)
#pragma unroll
        for (int v = 0; v < 8; ++v) pm[v] = fmaxf(pm[v], acc[t][v]);

    const int slot = wcol * 16 + n;
#pragma unroll
    for (int v = 0; v < 8; ++v) red[wrow * 16 + hi * 8 + v][slot] = pm[v];
    __syncthreads();
    if (tid < 32) {
        float m = red[tid][0];
        for (int s = 1; s < 64; ++s) m = fmaxf(m, red[tid][s]);
        rowmax[tid] = m;
    }
    __syncthreads();

    // --- exp + row sum ---
    float rm[8], ps[8];
#pragma unroll
    for (int v = 0; v < 8; ++v) {
        rm[v] = rowmax[wrow * 16 + hi * 8 + v];
        ps[v] = 0.f;
    }
#pragma unroll
    for (int t = 0; t < 32; ++t)
#pragma unroll
        for (int v = 0; v < 8; ++v) {
            float e = __expf(acc[t][v] - rm[v]);
            acc[t][v] = e;
            ps[v] += e;
        }
#pragma unroll
    for (int v = 0; v < 8; ++v) red[wrow * 16 + hi * 8 + v][slot] = ps[v];
    __syncthreads();
    if (tid < 32) {
        float s = 0.f;
        for (int k = 0; k < 64; ++k) s += red[tid][k];
        rowsum[tid] = s;
    }
    __syncthreads();

    // --- normalize + sum over this block's 32 atoms (race-free slots) ---
    float irs[8];
#pragma unroll
    for (int v = 0; v < 8; ++v) irs[v] = 1.0f / rowsum[wrow * 16 + hi * 8 + v];
#pragma unroll
    for (int t = 0; t < 32; ++t) {
        float s = 0.f;
#pragma unroll
        for (int v = 0; v < 8; ++v) s += acc[t][v] * irs[v];
        fpred[wcol * 512 + t * 16 + n][wrow * 2 + hi] = s;  // unique writer
    }
    __syncthreads();

    float* out = fppart + ((size_t)hb * 512 + b) * 2048;
    for (int o = tid; o < 2048; o += 256)
        out[o] += fpred[o][0] + fpred[o][1] + fpred[o][2] + fpred[o][3];
}

// ---------------------------------------------------------------------------
// K6: fp = fppart[0] + fppart[1]
// ---------------------------------------------------------------------------
__global__ void finalize_kernel(const float* __restrict__ fppart, float* __restrict__ fp) {
    int i = blockIdx.x * 256 + threadIdx.x;
    if (i < 512 * 2048) fp[i] = fppart[i] + fppart[512 * 2048 + i];
}

// ---------------------------------------------------------------------------
extern "C" void kernel_launch(void* const* d_in, const int* in_sizes, int n_in,
                              void* d_out, int out_size, void* d_ws, size_t ws_size,
                              hipStream_t stream) {
    const float* x        = (const float*)d_in[0];   // (512,64,64,40)
    const float* W_inner  = (const float*)d_in[1];   // (4,39,39)
    const float* b_inner  = (const float*)d_in[2];   // (4,1,39)
    const float* W_output = (const float*)d_in[3];   // (4,39,2048)
    const float* b_output = (const float*)d_in[4];   // (4,1,2048)
    float* fp = (float*)d_out;                       // (512,2048)

    char* w = (char*)d_ws;
    float*    attrs32 = (float*)w;     w += (size_t)512 * 64 * 40 * 4;
    float*    S       = (float*)w;     w += (size_t)512 * 64 * 40 * 4;
    float*    cbuf    = (float*)w;     w += (size_t)512 * 64 * 4;
    _Float16* attrs16 = (_Float16*)w;  w += (size_t)512 * 64 * 64 * 2;
    _Float16* woutT   = (_Float16*)w;  w += (size_t)4 * 2048 * 64 * 2;
    float*    fppart  = (float*)w;     w += (size_t)2 * 512 * 2048 * 4;

    zero_kernel<<<8192, 256, 0, stream>>>(fppart, 2 * 512 * 2048);
    woutT_kernel<<<2048, 256, 0, stream>>>(W_output, woutT);
    prep_kernel<<<512 * 64, 64, 0, stream>>>(x, attrs32, attrs16, S);
    colsum_kernel<<<512, 64, 0, stream>>>(x, cbuf);

    fp_kernel<<<1024, 256, 0, stream>>>(attrs16, woutT, b_output, fppart, 0);
    for (int d = 1; d <= 3; ++d) {
        depth_kernel<<<8192, 256, 0, stream>>>(W_inner, b_inner, S, attrs32,
                                               attrs16, cbuf, d);
        fp_kernel<<<1024, 256, 0, stream>>>(attrs16, woutT, b_output, fppart, d);
    }
    finalize_kernel<<<4096, 256, 0, stream>>>(fppart, fp);
}
